// GATNet_43980465111194
// MI455X (gfx1250) — matrix operand order repaired and checked by reading, not compiled, not run on hardware
//
#include <hip/hip_runtime.h>
#include <float.h>

typedef __attribute__((ext_vector_type(16))) __bf16 v16bf;
typedef __attribute__((ext_vector_type(8)))  __bf16 v8bf;
typedef __attribute__((ext_vector_type(8)))  float  v8f;

static inline int nblk(long long n, int t) { return (int)((n + t - 1) / t); }

__device__ __forceinline__ float leaky02(float x) { return x > 0.f ? x : 0.2f * x; }

// float atomic max via ordered-int trick (valid across mixed signs, init -FLT_MAX)
__device__ __forceinline__ void atomicMaxF(float* addr, float v) {
  if (v >= 0.f) atomicMax((int*)addr, __float_as_int(v));
  else          atomicMin((unsigned int*)addr, __float_as_uint(v));
}

__global__ void fill_f(float* __restrict__ p, long long n, float v) {
  long long i = (long long)blockIdx.x * blockDim.x + threadIdx.x;
  if (i < n) p[i] = v;
}

// ===================== WMMA GEMM (fp32 in/out, bf16 compute) =====================
// C[M,N] = act(A[M,K] @ B[K,N] + bias). A may be a piecewise concat of up to 4
// row-major fp32 sources along K (fuses the reference's concatenations).
struct APiece {
  const float *p0, *p1, *p2, *p3;
  int b0, b1, b2, b3;   // exclusive K boundaries
  int l0, l1, l2, l3;   // per-source row strides
};

static inline APiece mkplain(const float* p, int K, int ld) {
  APiece a; a.p0 = a.p1 = a.p2 = a.p3 = p;
  a.b0 = a.b1 = a.b2 = a.b3 = K;
  a.l0 = a.l1 = a.l2 = a.l3 = ld;
  return a;
}

// Branchless piecewise load: all candidate indices computed unconditionally,
// ternary selects (v_cndmask), one guarded load.
__device__ __forceinline__ float loadA_pw(const APiece& A, int row, int k, bool rowok) {
  size_t i0 = (size_t)row * A.l0 + k;
  size_t i1 = (size_t)row * A.l1 + (k - A.b0);
  size_t i2 = (size_t)row * A.l2 + (k - A.b1);
  size_t i3 = (size_t)row * A.l3 + (k - A.b2);
  size_t idx      = k < A.b0 ? i0   : k < A.b1 ? i1   : k < A.b2 ? i2   : i3;
  const float* p  = k < A.b0 ? A.p0 : k < A.b1 ? A.p1 : k < A.b2 ? A.p2 : A.p3;
  bool ok = rowok && (k < A.b3);
  return ok ? p[idx] : 0.f;   // zero-pad K to a multiple of 32
}

#define TM 128
#define TN 128
#define TK 32
// LDS row stride in bf16 elements: 2 fragment halves of 16 + 8 pad elements.
// 40*2 = 80 bytes => 20-bank stride (conflict-free), half offsets 0/32 B => 16B aligned.
#define LSTR 40

__device__ __forceinline__ v16bf ld_frag(const __bf16* p) {
  v8bf lo = *(const v8bf*)(p);
  v8bf hi = *(const v8bf*)(p + 8);
  return __builtin_shufflevector(lo, hi, 0, 1, 2, 3, 4, 5, 6, 7,
                                 8, 9, 10, 11, 12, 13, 14, 15);
}

template<bool RELU, bool PW>
__global__ __launch_bounds__(256) void gemm_wmma_bf16(
    APiece A, const float* __restrict__ Bw, const float* __restrict__ bias,
    float* __restrict__ C, int M, int N, int K, int ldb, int ldc)
{
  // Tiles stored pre-swizzled in WMMA fragment order:
  //  A fragment slot i of lane-half h <-> K = i + 16*h  (linear: offset == K)
  //  B fragment slot i of lane-half h <-> K = i + 16*h  (linear: offset == K)
  __shared__ __align__(32) __bf16 shA[TM * LSTR];
  __shared__ __align__(32) __bf16 shB[TN * LSTR];

  const int tid  = threadIdx.x;
  const int wid  = tid >> 5;
  const int lane = tid & 31;
  const int half = lane >> 4;
  const int l16  = lane & 15;
  const int wm   = wid >> 1;   // 0..3 : 32-row slab
  const int wn   = wid & 1;    // 0..1 : 64-col slab

  const int m0 = blockIdx.y * TM;
  const int n0 = blockIdx.x * TN;

  v8f acc[2][4];
#pragma unroll
  for (int mt = 0; mt < 2; ++mt)
#pragma unroll
    for (int nt = 0; nt < 4; ++nt)
#pragma unroll
      for (int j = 0; j < 8; ++j) acc[mt][nt][j] = 0.f;

  const int Kpad = ((K + TK - 1) / TK) * TK;
  for (int k0 = 0; k0 < Kpad; k0 += TK) {
    // stage A tile 128x32 (fp32 -> bf16), swizzled into fragment order
#pragma unroll
    for (int it = 0; it < (TM * TK) / 256; ++it) {
      int i = tid + it * 256;
      int r = i >> 5, kk = i & 31;
      int row = m0 + r;
      int kg = k0 + kk;
      float v;
      if (PW) v = loadA_pw(A, row, kg, row < M);
      else    v = (row < M && kg < K) ? A.p0[(size_t)row * A.l0 + kg] : 0.f;
      shA[r * LSTR + kk] = (__bf16)v;
    }
    // stage B tile 32x128, swizzled (offset within row == K)
#pragma unroll
    for (int it = 0; it < (TK * TN) / 256; ++it) {
      int i = tid + it * 256;
      int kk = i >> 7, c = i & 127;
      int kg = k0 + kk, col = n0 + c;
      float v = (kg < K && col < N) ? Bw[(size_t)kg * ldb + col] : 0.f;
      shB[c * LSTR + kk] = (__bf16)v;
    }
    __syncthreads();

    v16bf afr[2], bfr[4];
#pragma unroll
    for (int mt = 0; mt < 2; ++mt) {
      int r = wm * 32 + mt * 16 + l16;
      afr[mt] = ld_frag(shA + r * LSTR + half * 16);
    }
#pragma unroll
    for (int nt = 0; nt < 4; ++nt) {
      int c = wn * 64 + nt * 16 + l16;
      bfr[nt] = ld_frag(shB + c * LSTR + half * 16);
    }
#pragma unroll
    for (int mt = 0; mt < 2; ++mt)
#pragma unroll
      for (int nt = 0; nt < 4; ++nt)
        acc[mt][nt] = __builtin_amdgcn_wmma_f32_16x16x32_bf16(
            false, afr[mt], false, bfr[nt], (short)0, acc[mt][nt], false, false);
    __syncthreads();
  }

  // C 16x16 f32: lanes 0-15 -> M=j, lanes 16-31 -> M=8+j, N=l16
#pragma unroll
  for (int mt = 0; mt < 2; ++mt)
#pragma unroll
    for (int nt = 0; nt < 4; ++nt) {
      int col = n0 + wn * 64 + nt * 16 + l16;
      if (col >= N) continue;
      float bv = bias ? bias[col] : 0.f;
#pragma unroll
      for (int j = 0; j < 8; ++j) {
        int row = m0 + wm * 32 + mt * 16 + j + 8 * half;
        if (row < M) {
          float v = acc[mt][nt][j] + bv;
          if (RELU) v = fmaxf(v, 0.f);
          C[(size_t)row * ldc + col] = v;
        }
      }
    }
}

static void gemm_launch(hipStream_t st, const APiece& A, const float* B, const float* bias,
                        float* C, int M, int N, int K, int ldb, int ldc, bool relu) {
  dim3 grid((N + TN - 1) / TN, (M + TM - 1) / TM);
  bool pw = (A.b0 != A.b3);   // piecewise concat A?
  if (pw) {
    if (relu) gemm_wmma_bf16<true,  true ><<<grid, 256, 0, st>>>(A, B, bias, C, M, N, K, ldb, ldc);
    else      gemm_wmma_bf16<false, true ><<<grid, 256, 0, st>>>(A, B, bias, C, M, N, K, ldb, ldc);
  } else {
    if (relu) gemm_wmma_bf16<true,  false><<<grid, 256, 0, st>>>(A, B, bias, C, M, N, K, ldb, ldc);
    else      gemm_wmma_bf16<false, false><<<grid, 256, 0, st>>>(A, B, bias, C, M, N, K, ldb, ldc);
  }
}

// ===================== GAT edge machinery =====================
__device__ __forceinline__ void edge_sd(const int* __restrict__ ei, int E, int e, int& s, int& d) {
  if (e < E) { s = ei[e]; d = ei[E + e]; }
  else       { s = e - E; d = s; }          // implicit self-loops
}

__global__ void node_scores(const float* __restrict__ xp, const float* __restrict__ as,
                            const float* __restrict__ ad, float* __restrict__ es,
                            float* __restrict__ ed, int Nn, int H, int C) {
  long long i = (long long)blockIdx.x * blockDim.x + threadIdx.x;
  if (i >= (long long)Nn * H) return;
  int n = (int)(i / H), h = (int)(i % H);
  const float* row = xp + (size_t)n * H * C + (size_t)h * C;
  float s = 0.f, d = 0.f;
  for (int c = 0; c < C; ++c) { float v = row[c]; s += v * as[h * C + c]; d += v * ad[h * C + c]; }
  es[i] = s; ed[i] = d;
}

__global__ void edge_max_k(const int* __restrict__ ei, int E, int Nn,
                           const float* __restrict__ es, const float* __restrict__ ed,
                           float* __restrict__ mx, int H) {
  long long i = (long long)blockIdx.x * blockDim.x + threadIdx.x;
  long long tot = (long long)(E + Nn) * H;
  if (i >= tot) return;
  int e = (int)(i / H), h = (int)(i % H);
  int s, d; edge_sd(ei, E, e, s, d);
  float v = leaky02(es[(size_t)s * H + h] + ed[(size_t)d * H + h]);
  atomicMaxF(&mx[(size_t)d * H + h], v);
}

__global__ void edge_sum_k(const int* __restrict__ ei, int E, int Nn,
                           const float* __restrict__ es, const float* __restrict__ ed,
                           const float* __restrict__ mx, float* __restrict__ dn, int H) {
  long long i = (long long)blockIdx.x * blockDim.x + threadIdx.x;
  long long tot = (long long)(E + Nn) * H;
  if (i >= tot) return;
  int e = (int)(i / H), h = (int)(i % H);
  int s, d; edge_sd(ei, E, e, s, d);
  float v = leaky02(es[(size_t)s * H + h] + ed[(size_t)d * H + h]);
  atomicAdd(&dn[(size_t)d * H + h], expf(v - mx[(size_t)d * H + h]));
}

// one wave per (edge, head); lanes stride the channels
__global__ void edge_scatter_k(const int* __restrict__ ei, int E, int Nn,
                               const float* __restrict__ es, const float* __restrict__ ed,
                               const float* __restrict__ mx, const float* __restrict__ dn,
                               const float* __restrict__ xp, float* __restrict__ ob,
                               int H, int C) {
  long long wv = ((long long)blockIdx.x * blockDim.x + threadIdx.x) >> 5;
  int lane = threadIdx.x & 31;
  long long tot = (long long)(E + Nn) * H;
  if (wv >= tot) return;
  int e = (int)(wv / H), h = (int)(wv % H);
  int s, d; edge_sd(ei, E, e, s, d);
  float v = leaky02(es[(size_t)s * H + h] + ed[(size_t)d * H + h]);
  float alpha = expf(v - mx[(size_t)d * H + h]) / dn[(size_t)d * H + h];
  const float* sr = xp + (size_t)s * H * C + (size_t)h * C;
  float*       dr = ob + (size_t)d * H * C + (size_t)h * C;
  for (int c = lane; c < C; c += 32) atomicAdd(&dr[c], alpha * sr[c]);
}

template<bool ELU_ACT>
__global__ void finalize_k(float* __restrict__ o, const float* __restrict__ bias,
                           long long n, int HC) {
  long long i = (long long)blockIdx.x * blockDim.x + threadIdx.x;
  if (i >= n) return;
  float v = o[i] + bias[(int)(i % HC)];
  o[i] = ELU_ACT ? (v > 0.f ? v : (expf(v) - 1.f)) : fmaxf(v, 0.f);
}

__global__ void pool_max(const float* __restrict__ h, const int* __restrict__ batch,
                         float* __restrict__ g, int Nn, int C) {
  long long i = (long long)blockIdx.x * blockDim.x + threadIdx.x;
  if (i >= (long long)Nn * C) return;
  int n = (int)(i / C), c = (int)(i % C);
  atomicMaxF(&g[(size_t)batch[n] * C + c], h[i]);
}

// inputs are relu'd (>=0), so the reference's isfinite->0 fix == max(g,0)
__global__ void pool_fix(float* __restrict__ g, long long n) {
  long long i = (long long)blockIdx.x * blockDim.x + threadIdx.x;
  if (i < n) g[i] = fmaxf(g[i], 0.f);
}

// out = clip([h2|xtc] @ out_W + out_b, -100, 100); one wave per batch row
__global__ void final_out(const float* __restrict__ h2, const float* __restrict__ xtc,
                          const float* __restrict__ W, const float* __restrict__ b,
                          float* __restrict__ out, int Bn) {
  int wv = (int)(((long long)blockIdx.x * blockDim.x + threadIdx.x) >> 5);
  int lane = threadIdx.x & 31;
  if (wv >= Bn) return;
  float s = 0.f;
  for (int k = lane; k < 266; k += 32) {
    float x = (k < 256) ? h2[(size_t)wv * 256 + k] : xtc[(size_t)wv * 10 + (k - 256)];
    s += x * W[k];
  }
  for (int o = 16; o > 0; o >>= 1) s += __shfl_down(s, o, 32);
  if (lane == 0) {
    float r = s + b[0];
    out[wv] = fminf(fmaxf(r, -100.f), 100.f);
  }
}

// ===================== drug branch driver =====================
struct BranchParams {
  const float *W1, *as1, *ad1, *b1, *W2, *as2, *ad2, *b2, *fcW, *fcb;
};

static void run_branch(hipStream_t st, const float* x, const int* ei, const int* batch,
                       const BranchParams& P,
                       float* xp, float* ob, float* es, float* ed, float* mx, float* dn,
                       float* g, float* v_out, int Nn, int Ee, int Bb) {
  const int H1 = 10, C1 = 78, HC1 = 780, C2 = 128;
  const long long Et = (long long)Ee + Nn;

  // ---- layer 1: GATConv(78 -> 10x78), ELU ----
  gemm_launch(st, mkplain(x, 78, 78), P.W1, nullptr, xp, Nn, HC1, 78, HC1, HC1, false);
  node_scores<<<nblk((long long)Nn * H1, 256), 256, 0, st>>>(xp, P.as1, P.ad1, es, ed, Nn, H1, C1);
  fill_f<<<nblk((long long)Nn * H1, 256), 256, 0, st>>>(mx, (long long)Nn * H1, -FLT_MAX);
  fill_f<<<nblk((long long)Nn * H1, 256), 256, 0, st>>>(dn, (long long)Nn * H1, 0.f);
  fill_f<<<nblk((long long)Nn * HC1, 256), 256, 0, st>>>(ob, (long long)Nn * HC1, 0.f);
  edge_max_k<<<nblk(Et * H1, 256), 256, 0, st>>>(ei, Ee, Nn, es, ed, mx, H1);
  edge_sum_k<<<nblk(Et * H1, 256), 256, 0, st>>>(ei, Ee, Nn, es, ed, mx, dn, H1);
  edge_scatter_k<<<nblk(Et * H1 * 32, 256), 256, 0, st>>>(ei, Ee, Nn, es, ed, mx, dn, xp, ob, H1, C1);
  finalize_k<true><<<nblk((long long)Nn * HC1, 256), 256, 0, st>>>(ob, P.b1, (long long)Nn * HC1, HC1);

  // ---- layer 2: GATConv(780 -> 128), ReLU (xp buffer reused) ----
  float* xp2  = xp;
  float* out2 = xp + (size_t)Nn * C2;
  gemm_launch(st, mkplain(ob, 780, 780), P.W2, nullptr, xp2, Nn, C2, 780, C2, C2, false);
  node_scores<<<nblk(Nn, 256), 256, 0, st>>>(xp2, P.as2, P.ad2, es, ed, Nn, 1, C2);
  fill_f<<<nblk(Nn, 256), 256, 0, st>>>(mx, Nn, -FLT_MAX);
  fill_f<<<nblk(Nn, 256), 256, 0, st>>>(dn, Nn, 0.f);
  fill_f<<<nblk((long long)Nn * C2, 256), 256, 0, st>>>(out2, (long long)Nn * C2, 0.f);
  edge_max_k<<<nblk(Et, 256), 256, 0, st>>>(ei, Ee, Nn, es, ed, mx, 1);
  edge_sum_k<<<nblk(Et, 256), 256, 0, st>>>(ei, Ee, Nn, es, ed, mx, dn, 1);
  edge_scatter_k<<<nblk(Et * 32, 256), 256, 0, st>>>(ei, Ee, Nn, es, ed, mx, dn, xp2, out2, 1, C2);
  finalize_k<false><<<nblk((long long)Nn * C2, 256), 256, 0, st>>>(out2, P.b2, (long long)Nn * C2, C2);

  // ---- global max pool + fcg ----
  fill_f<<<nblk((long long)Bb * C2, 256), 256, 0, st>>>(g, (long long)Bb * C2, -FLT_MAX);
  pool_max<<<nblk((long long)Nn * C2, 256), 256, 0, st>>>(out2, batch, g, Nn, C2);
  pool_fix<<<nblk((long long)Bb * C2, 256), 256, 0, st>>>(g, (long long)Bb * C2);
  gemm_launch(st, mkplain(g, C2, C2), P.fcW, P.fcb, v_out, Bb, C2, C2, C2, C2, true);
}

// ===================== entry =====================
extern "C" void kernel_launch(void* const* d_in, const int* in_sizes, int n_in,
                              void* d_out, int out_size, void* d_ws, size_t ws_size,
                              hipStream_t stream) {
  const int Nn = in_sizes[0] / 78;     // 30000
  const int Ee = in_sizes[1] / 2;      // 120000
  const int Bb = in_sizes[9] / 10;     // 2048
  const int C1c = 23808, C2c = 3171, C3c = 627, NTC = 10;

  const float* xd1 = (const float*)d_in[0];
  const int*   ei1 = (const int*)  d_in[1];
  const int*   bt1 = (const int*)  d_in[2];
  const float* xd2 = (const float*)d_in[3];
  const int*   ei2 = (const int*)  d_in[4];
  const int*   bt2 = (const int*)  d_in[5];
  const float* xc1 = (const float*)d_in[6];
  const float* xc2 = (const float*)d_in[7];
  const float* xc3 = (const float*)d_in[8];
  const float* xtc = (const float*)d_in[9];
  BranchParams P1 = { (const float*)d_in[10], (const float*)d_in[11], (const float*)d_in[12],
                      (const float*)d_in[13], (const float*)d_in[14], (const float*)d_in[15],
                      (const float*)d_in[16], (const float*)d_in[17], (const float*)d_in[18],
                      (const float*)d_in[19] };
  BranchParams P2 = { (const float*)d_in[20], (const float*)d_in[21], (const float*)d_in[22],
                      (const float*)d_in[23], (const float*)d_in[24], (const float*)d_in[25],
                      (const float*)d_in[26], (const float*)d_in[27], (const float*)d_in[28],
                      (const float*)d_in[29] };
  const float* clW1 = (const float*)d_in[30];
  const float* clb1 = (const float*)d_in[31];
  const float* clW2 = (const float*)d_in[32];
  const float* clb2 = (const float*)d_in[33];
  const float* f1W  = (const float*)d_in[34];
  const float* f1b  = (const float*)d_in[35];
  const float* f2W  = (const float*)d_in[36];
  const float* f2b  = (const float*)d_in[37];
  const float* oW   = (const float*)d_in[38];
  const float* ob_  = (const float*)d_in[39];

  // workspace carve-up (floats)
  float* ws = (float*)d_ws;
  size_t o = 0;
  float* xp   = ws + o; o += (size_t)Nn * 780;
  float* outb = ws + o; o += (size_t)Nn * 780;
  float* es   = ws + o; o += (size_t)Nn * 10;
  float* ed   = ws + o; o += (size_t)Nn * 10;
  float* mx   = ws + o; o += (size_t)Nn * 10;
  float* dn   = ws + o; o += (size_t)Nn * 10;
  float* v1   = ws + o; o += (size_t)Bb * 128;
  float* v2   = ws + o; o += (size_t)Bb * 128;
  float* g    = ws + o; o += (size_t)Bb * 128;
  float* clh  = ws + o; o += (size_t)Bb * 2048;   // also reused as fc1 hidden
  float* clo  = ws + o; o += (size_t)Bb * 256;
  float* fh2  = ws + o; o += (size_t)Bb * 256;
  (void)ws_size; (void)n_in; (void)out_size;

  // two drug branches (sequential, shared big buffers)
  run_branch(stream, xd1, ei1, bt1, P1, xp, outb, es, ed, mx, dn, g, v1, Nn, Ee, Bb);
  run_branch(stream, xd2, ei2, bt2, P2, xp, outb, es, ed, mx, dn, g, v2, Nn, Ee, Bb);

  // cell-line MLP: fused concat([xc1,xc2,xc3,xtc]) @ cl_W1 (K = 27616 = 863*32)
  APiece Acl;
  Acl.p0 = xc1; Acl.p1 = xc2; Acl.p2 = xc3; Acl.p3 = xtc;
  Acl.b0 = C1c; Acl.b1 = C1c + C2c; Acl.b2 = C1c + C2c + C3c; Acl.b3 = C1c + C2c + C3c + NTC;
  Acl.l0 = C1c; Acl.l1 = C2c; Acl.l2 = C3c; Acl.l3 = NTC;
  gemm_launch(stream, Acl, clW1, clb1, clh, Bb, 2048, 27616, 2048, 2048, true);
  gemm_launch(stream, mkplain(clh, 2048, 2048), clW2, clb2, clo, Bb, 256, 2048, 256, 256, true);

  // fc stack: fused concat([v1, v2, xcl, xtc]) @ fc1_W (K = 522, zero-padded to 544)
  APiece Afc;
  Afc.p0 = v1; Afc.p1 = v2; Afc.p2 = clo; Afc.p3 = xtc;
  Afc.b0 = 128; Afc.b1 = 256; Afc.b2 = 512; Afc.b3 = 522;
  Afc.l0 = 128; Afc.l1 = 128; Afc.l2 = 256; Afc.l3 = NTC;
  float* fh1 = clh;   // reuse, cl hidden is dead now
  gemm_launch(stream, Afc, f1W, f1b, fh1, Bb, 2048, 522, 2048, 2048, true);
  gemm_launch(stream, mkplain(fh1, 2048, 2048), f2W, f2b, fh2, Bb, 256, 2048, 256, 256, true);

  // output head: [h2 | xtc] @ out_W + out_b, clipped
  final_out<<<nblk((long long)Bb * 32, 256), 256, 0, stream>>>(fh2, xtc, oW, ob_, (float*)d_out, Bb);
}
